// GLFMSABlock_56985626083962
// MI455X (gfx1250) — matrix operand order repaired and checked by reading, not compiled, not run on hardware
//
#include <hip/hip_runtime.h>
#include <hip/hip_bf16.h>
#include <stdint.h>

// ---------------- problem constants ----------------
constexpr int kD  = 512;
constexpr int kS  = 2048;
constexpr int kB  = 4;
constexpr int kH  = 8;     // heads (global and local)
constexpr int kDH = 64;    // head dim
constexpr int kM  = kB * kS;   // 8192 rows

#define ACT_NONE 0
#define ACT_GELU 1

// ---------------- WMMA fragment types ----------------
typedef __attribute__((ext_vector_type(16))) _Float16 v16h;
typedef __attribute__((ext_vector_type(8)))  _Float16 v8h;
typedef __attribute__((ext_vector_type(8)))  float    v8f;

union HV { v16h h; v8h h2[2]; };

static __device__ inline v8f zero8() {
    v8f z;
#pragma unroll
    for (int i = 0; i < 8; ++i) z[i] = 0.f;
    return z;
}

// ---------------- f32 -> f16 convert ----------------
__global__ void f32_to_f16_kernel(const float* __restrict__ in,
                                  _Float16* __restrict__ out, int n) {
    int i = blockIdx.x * blockDim.x + threadIdx.x;
    if (i < n) out[i] = (_Float16)in[i];
}

// ---------------- LayerNorm (optional residual, f32/f16 outs) -------------
// one block (256 threads) per row of D=512
__global__ void ln_kernel(const float* __restrict__ in, const float* __restrict__ res,
                          const float* __restrict__ g, const float* __restrict__ bta,
                          float* __restrict__ outF, _Float16* __restrict__ outH) {
    __shared__ float red[256];
    const int row = blockIdx.x;
    const int tid = threadIdx.x;
    const float* x = in + (size_t)row * kD;
    float v0 = x[tid];
    float v1 = x[tid + 256];
    if (res) {
        const float* rr = res + (size_t)row * kD;
        v0 += rr[tid];
        v1 += rr[tid + 256];
    }
    red[tid] = v0 + v1;
    __syncthreads();
    for (int o = 128; o > 0; o >>= 1) {
        if (tid < o) red[tid] += red[tid + o];
        __syncthreads();
    }
    float mean = red[0] * (1.f / kD);
    __syncthreads();
    float c0 = v0 - mean, c1 = v1 - mean;
    red[tid] = c0 * c0 + c1 * c1;
    __syncthreads();
    for (int o = 128; o > 0; o >>= 1) {
        if (tid < o) red[tid] += red[tid + o];
        __syncthreads();
    }
    float rstd = rsqrtf(red[0] * (1.f / kD) + 1e-5f);
    float y0 = c0 * rstd * g[tid] + bta[tid];
    float y1 = c1 * rstd * g[tid + 256] + bta[tid + 256];
    size_t base = (size_t)row * kD;
    if (outF) { outF[base + tid] = y0; outF[base + tid + 256] = y1; }
    if (outH) { outH[base + tid] = (_Float16)y0; outH[base + tid + 256] = (_Float16)y1; }
}

// ---------------- generic WMMA GEMM, 32x64 C tile per wave ----------------
// C[M,N] = A[M,K] * W[N,K]^T + bias;  A,W f16 row-major; out f32 and/or f16.
// block: 256 threads = 8 waves as 2 (M) x 4 (N); block tile 64 x 256.
// grid: (M/64, N/256). Fragments loaded straight from global, 16 contiguous K per lane,
// software-pipelined one K-step ahead so loads overlap the WMMA chain.
struct Frags { HV a0, a1, b0, b1, b2, b3; };

static __device__ inline void load_frags(Frags& f,
                                         const _Float16* pa0, const _Float16* pa1,
                                         const _Float16* pb0, const _Float16* pb1,
                                         const _Float16* pb2, const _Float16* pb3,
                                         int k0) {
    f.a0.h2[0] = *(const v8h*)(pa0 + k0);
    f.a0.h2[1] = *(const v8h*)(pa0 + k0 + 8);
    f.a1.h2[0] = *(const v8h*)(pa1 + k0);
    f.a1.h2[1] = *(const v8h*)(pa1 + k0 + 8);
    f.b0.h2[0] = *(const v8h*)(pb0 + k0);
    f.b0.h2[1] = *(const v8h*)(pb0 + k0 + 8);
    f.b1.h2[0] = *(const v8h*)(pb1 + k0);
    f.b1.h2[1] = *(const v8h*)(pb1 + k0 + 8);
    f.b2.h2[0] = *(const v8h*)(pb2 + k0);
    f.b2.h2[1] = *(const v8h*)(pb2 + k0 + 8);
    f.b3.h2[0] = *(const v8h*)(pb3 + k0);
    f.b3.h2[1] = *(const v8h*)(pb3 + k0 + 8);
}

__global__ void gemm_wmma_kernel(const _Float16* __restrict__ A,
                                 const _Float16* __restrict__ W,
                                 const float* __restrict__ bias,
                                 float* __restrict__ outF,
                                 _Float16* __restrict__ outH,
                                 int N, int K, int act) {
    const int lane  = threadIdx.x & 31;
    const int wave  = threadIdx.x >> 5;
    const int wr    = wave >> 2;                    // 0..1
    const int wc    = wave & 3;                     // 0..3
    const int mbase = blockIdx.x * 64 + wr * 32;    // two 16-row tiles
    const int nbase = blockIdx.y * 256 + wc * 64;   // four 16-col tiles

    v8f acc[2][4];
#pragma unroll
    for (int mi = 0; mi < 2; ++mi)
#pragma unroll
        for (int ni = 0; ni < 4; ++ni) acc[mi][ni] = zero8();

    // per-lane fragment base addresses (A and B both 16 contiguous K per lane)
    const int am  = lane & 15;
    const int kqa = (lane >> 4) << 4;   // A: 16 contiguous K per lane
    const int kqb = (lane >> 4) << 4;   // B: 16 contiguous K per lane
    const _Float16* pa0 = A + (size_t)(mbase + am) * K + kqa;
    const _Float16* pa1 = pa0 + (size_t)16 * K;
    const _Float16* pb0 = W + (size_t)(nbase + am) * K + kqb;
    const _Float16* pb1 = pb0 + (size_t)16 * K;
    const _Float16* pb2 = pb0 + (size_t)32 * K;
    const _Float16* pb3 = pb0 + (size_t)48 * K;

    Frags cur, nxt;
    load_frags(cur, pa0, pa1, pb0, pb1, pb2, pb3, 0);

    for (int k0 = 0; k0 < K; k0 += 32) {
        const int kn = (k0 + 32 < K) ? (k0 + 32) : 0;  // wrap: data unused on last iter
        __builtin_prefetch(pa0 + k0 + 64, 0, 1);
        __builtin_prefetch(pb0 + k0 + 64, 0, 1);
        __builtin_prefetch(pb2 + k0 + 64, 0, 1);
        // issue next iteration's loads before consuming current fragments
        load_frags(nxt, pa0, pa1, pb0, pb1, pb2, pb3, kn);

        acc[0][0] = __builtin_amdgcn_wmma_f32_16x16x32_f16(false, cur.a0.h, false, cur.b0.h, (short)0, acc[0][0], false, false);
        acc[1][0] = __builtin_amdgcn_wmma_f32_16x16x32_f16(false, cur.a1.h, false, cur.b0.h, (short)0, acc[1][0], false, false);
        acc[0][1] = __builtin_amdgcn_wmma_f32_16x16x32_f16(false, cur.a0.h, false, cur.b1.h, (short)0, acc[0][1], false, false);
        acc[1][1] = __builtin_amdgcn_wmma_f32_16x16x32_f16(false, cur.a1.h, false, cur.b1.h, (short)0, acc[1][1], false, false);
        acc[0][2] = __builtin_amdgcn_wmma_f32_16x16x32_f16(false, cur.a0.h, false, cur.b2.h, (short)0, acc[0][2], false, false);
        acc[1][2] = __builtin_amdgcn_wmma_f32_16x16x32_f16(false, cur.a1.h, false, cur.b2.h, (short)0, acc[1][2], false, false);
        acc[0][3] = __builtin_amdgcn_wmma_f32_16x16x32_f16(false, cur.a0.h, false, cur.b3.h, (short)0, acc[0][3], false, false);
        acc[1][3] = __builtin_amdgcn_wmma_f32_16x16x32_f16(false, cur.a1.h, false, cur.b3.h, (short)0, acc[1][3], false, false);

        cur = nxt;
    }

    // epilogue: bias + optional GELU, f32/f16 stores
#pragma unroll
    for (int mi = 0; mi < 2; ++mi) {
        const int rbase = mbase + mi * 16 + ((lane >> 4) << 3);
#pragma unroll
        for (int ni = 0; ni < 4; ++ni) {
            const int nc = nbase + ni * 16 + (lane & 15);
            const float bval = bias ? bias[nc] : 0.f;
#pragma unroll
            for (int r = 0; r < 8; ++r) {
                float v = acc[mi][ni][r] + bval;
                if (act == ACT_GELU)
                    v = 0.5f * v * (1.f + erff(v * 0.70710678118654752f));
                size_t o = (size_t)(rbase + r) * N + nc;
                if (outF) outF[o] = v;
                if (outH) outH[o] = (_Float16)v;
            }
        }
    }
}

// ---------------- global causal flash attention ----------------
// qkv: [B, S, 3D] f16 (q|k|v). One wave per (b, h, 16-query tile).
__global__ void flash_attn_kernel(const _Float16* __restrict__ qkv,
                                  _Float16* __restrict__ outH) {
    __shared__ _Float16 pbuf[8][16 * 32];
    const int lane = threadIdx.x & 31;
    const int wave = threadIdx.x >> 5;
    const int gw = blockIdx.x * 8 + wave;
    const int qt = gw & (kS / 16 - 1);  // 0..127
    const int bh = gw >> 7;
    const int h  = bh & (kH - 1);
    const int b  = bh >> 3;

    const size_t rs = 3 * kD;
    const _Float16* qbase = qkv + (size_t)b * kS * rs + h * kDH;
    const _Float16* kbase = qbase + kD;
    const _Float16* vbase = qbase + 2 * kD;

    // Q fragments for dh slices [0,32) and [32,64)
    HV qf[2];
    {
        const int qm = qt * 16 + (lane & 15);
        const int kq = (lane >> 4) << 4;
        const _Float16* p = qbase + (size_t)qm * rs + kq;
        qf[0].h2[0] = *(const v8h*)(p);
        qf[0].h2[1] = *(const v8h*)(p + 8);
        qf[1].h2[0] = *(const v8h*)(p + 32);
        qf[1].h2[1] = *(const v8h*)(p + 40);
    }

    v8f o[4];
#pragma unroll
    for (int t = 0; t < 4; ++t) o[t] = zero8();
    float mrow[8], lrow[8];
#pragma unroll
    for (int r = 0; r < 8; ++r) { mrow[r] = -1e30f; lrow[r] = 0.f; }

    const int qhi = qt * 16 + 15;
    for (int kc = 0; kc * 32 <= qhi; ++kc) {
        const int kb = kc * 32;
        v8f s[2];
#pragma unroll
        for (int t = 0; t < 2; ++t) {
            v8f sc = zero8();
#pragma unroll
            for (int ks = 0; ks < 2; ++ks) {
                HV kf;
                const int kn = kb + t * 16 + (lane & 15);
                const int kq = ks * 32 + ((lane >> 4) << 4);
                const _Float16* p = kbase + (size_t)kn * rs + kq;
                kf.h2[0] = *(const v8h*)(p);
                kf.h2[1] = *(const v8h*)(p + 8);
                sc = __builtin_amdgcn_wmma_f32_16x16x32_f16(false, qf[ks].h, false, kf.h,
                                                            (short)0, sc, false, false);
            }
            s[t] = sc;
        }

        // mask + online softmax (row groups: lanes 0-15 -> rows 0-7, 16-31 -> 8-15)
        float scale[8];
#pragma unroll
        for (int r = 0; r < 8; ++r) {
            const int qm  = qt * 16 + r + ((lane >> 4) << 3);
            const int k0c = kb + (lane & 15);
            float a0 = s[0][r] * 0.125f;
            float a1 = s[1][r] * 0.125f;
            if (k0c > qm)      a0 = -1e30f;
            if (k0c + 16 > qm) a1 = -1e30f;
            float mx = fmaxf(a0, a1);
            mx = fmaxf(mx, __shfl_xor(mx, 1, 32));
            mx = fmaxf(mx, __shfl_xor(mx, 2, 32));
            mx = fmaxf(mx, __shfl_xor(mx, 4, 32));
            mx = fmaxf(mx, __shfl_xor(mx, 8, 32));
            float nm  = fmaxf(mrow[r], mx);
            float sc2 = __expf(mrow[r] - nm);
            float p0 = __expf(a0 - nm);
            float p1 = __expf(a1 - nm);
            float rsum = p0 + p1;
            rsum += __shfl_xor(rsum, 1, 32);
            rsum += __shfl_xor(rsum, 2, 32);
            rsum += __shfl_xor(rsum, 4, 32);
            rsum += __shfl_xor(rsum, 8, 32);
            lrow[r] = lrow[r] * sc2 + rsum;
            mrow[r] = nm;
            scale[r] = sc2;
            s[0][r] = p0;
            s[1][r] = p1;
        }
#pragma unroll
        for (int r = 0; r < 8; ++r) {
            o[0][r] *= scale[r]; o[1][r] *= scale[r];
            o[2][r] *= scale[r]; o[3][r] *= scale[r];
        }

        // C-layout P -> LDS (row-major 16x32 f16, per-wave region) -> A-layout
        _Float16* pw = pbuf[wave];
#pragma unroll
        for (int r = 0; r < 8; ++r) {
            const int ml = r + ((lane >> 4) << 3);
            pw[ml * 32 + (lane & 15)]      = (_Float16)s[0][r];
            pw[ml * 32 + 16 + (lane & 15)] = (_Float16)s[1][r];
        }
        __asm__ volatile("s_wait_dscnt 0" ::: "memory");
        HV pf;
        {
            const int am = lane & 15;
            const int kq = (lane >> 4) << 4;
            const _Float16* pp = &pw[am * 32 + kq];
            pf.h2[0] = *(const v8h*)(pp);
            pf.h2[1] = *(const v8h*)(pp + 8);
        }

        // O += P[16x32] x V[32x64]
#pragma unroll
        for (int ot = 0; ot < 4; ++ot) {
            HV vf;
            const int dn = ot * 16 + (lane & 15);
            const int kk = (lane >> 4) << 4;
#pragma unroll
            for (int hh = 0; hh < 16; ++hh)
                vf.h[hh] = vbase[(size_t)(kb + kk + hh) * rs + dn];
            o[ot] = __builtin_amdgcn_wmma_f32_16x16x32_f16(false, pf.h, false, vf.h,
                                                           (short)0, o[ot], false, false);
        }
    }

    // normalize + store f16
    const int dlo = lane & 15;
#pragma unroll
    for (int r = 0; r < 8; ++r) {
        float inv = 1.f / lrow[r];
        const int qm = qt * 16 + r + ((lane >> 4) << 3);
        size_t base = ((size_t)b * kS + qm) * kD + h * kDH;
#pragma unroll
        for (int ot = 0; ot < 4; ++ot)
            outH[base + ot * 16 + dlo] = (_Float16)(o[ot][r] * inv);
    }
}

// ---------------- local windowed attention (WIN=2) ----------------
// one thread per (b, s, head)
__global__ void local_attn_kernel(const _Float16* __restrict__ qkv,
                                  _Float16* __restrict__ outH) {
    int t = blockIdx.x * blockDim.x + threadIdx.x;
    if (t >= kB * kS * kH) return;
    const int h = t & (kH - 1);
    const int s = (t >> 3) & (kS - 1);
    const int b = t >> 14;
    const int s0 = s & ~1;
    const size_t rs = 3 * kD;
    const _Float16* q  = qkv + ((size_t)b * kS + s) * rs + h * kDH;
    const _Float16* k0 = qkv + ((size_t)b * kS + s0) * rs + kD + h * kDH;
    const _Float16* k1 = k0 + rs;
    const _Float16* v0 = k0 + kD;
    const _Float16* v1 = k1 + kD;
    float d0 = 0.f, d1 = 0.f;
    for (int d = 0; d < kDH; ++d) {
        float qd = (float)q[d];
        d0 += qd * (float)k0[d];
        d1 += qd * (float)k1[d];
    }
    d0 *= 0.125f; d1 *= 0.125f;
    float mx = fmaxf(d0, d1);
    float e0 = __expf(d0 - mx), e1 = __expf(d1 - mx);
    float inv = 1.f / (e0 + e1);
    e0 *= inv; e1 *= inv;
    _Float16* o = outH + ((size_t)b * kS + s) * kD + h * kDH;
    for (int d = 0; d < kDH; ++d)
        o[d] = (_Float16)(e0 * (float)v0[d] + e1 * (float)v1[d]);
}

// ---------------- gated combine ----------------
__global__ void combine_kernel(const float* __restrict__ x, const float* __restrict__ gl,
                               const float* __restrict__ go, const float* __restrict__ lo,
                               float* __restrict__ out, int n) {
    int i = blockIdx.x * blockDim.x + threadIdx.x;
    if (i < n) {
        float gt = 1.f / (1.f + __expf(-gl[i]));
        out[i] = x[i] + gt * go[i] + (1.f - gt) * lo[i];
    }
}

// ---------------- launch ----------------
static inline size_t al256(size_t x) { return (x + 255) & ~(size_t)255; }

extern "C" void kernel_launch(void* const* d_in, const int* in_sizes, int n_in,
                              void* d_out, int out_size, void* d_ws, size_t ws_size,
                              hipStream_t stream) {
    const float* x      = (const float*)d_in[0];
    const float* ln1_g  = (const float*)d_in[1];
    const float* ln1_b  = (const float*)d_in[2];
    const float* ln2_g  = (const float*)d_in[3];
    const float* ln2_b  = (const float*)d_in[4];
    const float* ln3_g  = (const float*)d_in[5];
    const float* ln3_b  = (const float*)d_in[6];
    const float* gw_in  = (const float*)d_in[7];
    const float* gb_in  = (const float*)d_in[8];
    const float* gw_out = (const float*)d_in[9];
    const float* gb_out = (const float*)d_in[10];
    const float* lw_in  = (const float*)d_in[11];
    const float* lb_in  = (const float*)d_in[12];
    const float* lw_out = (const float*)d_in[13];
    const float* lb_out = (const float*)d_in[14];
    const float* gate_w = (const float*)d_in[15];
    const float* gate_b = (const float*)d_in[16];
    const float* ff_w1  = (const float*)d_in[17];
    const float* ff_b1  = (const float*)d_in[18];
    const float* ff_w2  = (const float*)d_in[19];
    const float* ff_b2  = (const float*)d_in[20];

    char* ws = (char*)d_ws;
    size_t off = 0;
    auto alloc = [&](size_t bytes) { void* p = ws + off; off = al256(off + bytes); return p; };

    const size_t nW_in  = (size_t)3 * kD * kD;
    const size_t nW_sq  = (size_t)kD * kD;
    const size_t nW_ff  = (size_t)4 * kD * kD;
    const size_t nTok   = (size_t)kM * kD;

    _Float16* gw_in_h  = (_Float16*)alloc(nW_in * 2);
    _Float16* lw_in_h  = (_Float16*)alloc(nW_in * 2);
    _Float16* gw_out_h = (_Float16*)alloc(nW_sq * 2);
    _Float16* lw_out_h = (_Float16*)alloc(nW_sq * 2);
    _Float16* gate_w_h = (_Float16*)alloc(nW_sq * 2);
    _Float16* ff_w1_h  = (_Float16*)alloc(nW_ff * 2);
    _Float16* ff_w2_h  = (_Float16*)alloc(nW_ff * 2);
    _Float16* xn_h     = (_Float16*)alloc(nTok * 2);
    _Float16* g_qkv_h  = (_Float16*)alloc(nTok * 3 * 2);
    _Float16* l_qkv_h  = (_Float16*)alloc(nTok * 3 * 2);
    _Float16* g_att_h  = (_Float16*)alloc(nTok * 2);
    _Float16* l_att_h  = (_Float16*)alloc(nTok * 2);
    _Float16* h2_h     = (_Float16*)alloc(nTok * 2);
    _Float16* ff1_h    = (_Float16*)alloc(nTok * 4 * 2);
    float* gate_f  = (float*)alloc(nTok * 4);
    float* g_out_f = (float*)alloc(nTok * 4);
    float* l_out_f = (float*)alloc(nTok * 4);
    float* x1_f    = (float*)alloc(nTok * 4);
    float* ff2_f   = (float*)alloc(nTok * 4);

    auto conv = [&](const float* src, _Float16* dst, size_t n) {
        f32_to_f16_kernel<<<dim3((unsigned)((n + 255) / 256)), dim3(256), 0, stream>>>(src, dst, (int)n);
    };
    conv(gw_in,  gw_in_h,  nW_in);
    conv(lw_in,  lw_in_h,  nW_in);
    conv(gw_out, gw_out_h, nW_sq);
    conv(lw_out, lw_out_h, nW_sq);
    conv(gate_w, gate_w_h, nW_sq);
    conv(ff_w1,  ff_w1_h,  nW_ff);
    conv(ff_w2,  ff_w2_h,  nW_ff);

    // LN1 -> f16 activations
    ln_kernel<<<dim3(kM), dim3(256), 0, stream>>>(x, nullptr, ln1_g, ln1_b, nullptr, xn_h);

    const dim3 blk(256);
    // QKV projections
    gemm_wmma_kernel<<<dim3(kM / 64, 3 * kD / 256), blk, 0, stream>>>(
        xn_h, gw_in_h, gb_in, nullptr, g_qkv_h, 3 * kD, kD, ACT_NONE);
    gemm_wmma_kernel<<<dim3(kM / 64, 3 * kD / 256), blk, 0, stream>>>(
        xn_h, lw_in_h, lb_in, nullptr, l_qkv_h, 3 * kD, kD, ACT_NONE);
    // gate linear (f32 out; sigmoid fused into combine)
    gemm_wmma_kernel<<<dim3(kM / 64, kD / 256), blk, 0, stream>>>(
        xn_h, gate_w_h, gate_b, gate_f, nullptr, kD, kD, ACT_NONE);

    // attention
    flash_attn_kernel<<<dim3(kB * kH * (kS / 16) / 8), blk, 0, stream>>>(g_qkv_h, g_att_h);
    local_attn_kernel<<<dim3(kB * kS * kH / 256), blk, 0, stream>>>(l_qkv_h, l_att_h);

    // out projections
    gemm_wmma_kernel<<<dim3(kM / 64, kD / 256), blk, 0, stream>>>(
        g_att_h, gw_out_h, gb_out, g_out_f, nullptr, kD, kD, ACT_NONE);
    gemm_wmma_kernel<<<dim3(kM / 64, kD / 256), blk, 0, stream>>>(
        l_att_h, lw_out_h, lb_out, l_out_f, nullptr, kD, kD, ACT_NONE);

    // gated residual combine
    combine_kernel<<<dim3((unsigned)(nTok / 256)), blk, 0, stream>>>(
        x, gate_f, g_out_f, l_out_f, x1_f, (int)nTok);

    // FFN
    ln_kernel<<<dim3(kM), dim3(256), 0, stream>>>(x1_f, nullptr, ln2_g, ln2_b, nullptr, h2_h);
    gemm_wmma_kernel<<<dim3(kM / 64, 4 * kD / 256), blk, 0, stream>>>(
        h2_h, ff_w1_h, ff_b1, nullptr, ff1_h, 4 * kD, kD, ACT_GELU);
    gemm_wmma_kernel<<<dim3(kM / 64, kD / 256), blk, 0, stream>>>(
        ff1_h, ff_w2_h, ff_b2, ff2_f, nullptr, kD, 4 * kD, ACT_NONE);

    // final residual + LN3 -> f32 output
    ln_kernel<<<dim3(kM), dim3(256), 0, stream>>>(x1_f, ff2_f, ln3_g, ln3_b, (float*)d_out, nullptr);
}
